// AttnNetwork_85779086836246
// MI455X (gfx1250) — matrix operand order repaired and checked by reading, not compiled, not run on hardware
//
#include <hip/hip_runtime.h>
#include <stdint.h>

// ---------------- problem constants (match reference) ----------------
#define BATCH   64
#define SEQ     50
#define TM1     49
#define WDIM    300
#define WDIMP   320     // padded to multiple of 32 (WMMA K)
#define HID     500
#define HIDP    512
#define GATE    2000    // 4*HID
#define VLAY    500
#define VLAYP   512
#define FEATK   1000
#define FEATKP  1024
#define VEN     32000
#define NTILES_V (VEN / 64)   // 500
#define MROWS   (BATCH * TM1) // 3136
#define PAD_TOKEN 1

typedef __attribute__((ext_vector_type(16))) _Float16 v16h;
typedef __attribute__((ext_vector_type(8)))  float    v8f;

#define LDSS 40   // LDS tile row stride in halfs (80B -> conflict-free for 16-lane frag reads)

// ---------------- CDNA5 async global->LDS copy helpers ----------------
__device__ __forceinline__ unsigned lds_off_of(const void* p) {
    return (unsigned)(unsigned long long)(const __attribute__((address_space(3))) char*)p;
}
// GLOBAL_LOAD_ASYNC_TO_LDS_B128, GVS form: mem = SADDR(u64) + VADDR(i32); dest = LDS_BASE + VDST
__device__ __forceinline__ void async_copy_b128(unsigned lds_off, const void* sbase, unsigned goff) {
    asm volatile("global_load_async_to_lds_b128 %0, %1, %2"
                 :: "v"(lds_off), "v"(goff), "s"(sbase)
                 : "memory");
}
__device__ __forceinline__ void wait_async0() {
    asm volatile("s_wait_asynccnt 0x0" ::: "memory");
}

// ---------------- WMMA fragment helpers ----------------
// A fragment (16x32 f16, MxK): lane l: m = l&15, khalf = l>>4.
//   elements 0..7  = K [ 8*khalf .. 8*khalf+7 ], elements 8..15 = K [ 16+8*khalf .. +7 ]
__device__ __forceinline__ v16h load_fragA(const _Float16* tile, int lane) {
    int m = lane & 15;
    int kh = (lane >> 4) & 1;
    const _Float16* p = tile + m * LDSS + kh * 8;
    v16h r;
    uint4* rp = reinterpret_cast<uint4*>(&r);
    rp[0] = *reinterpret_cast<const uint4*>(p);
    rp[1] = *reinterpret_cast<const uint4*>(p + 16);
    return r;
}
// B fragment (32x16 f16, KxN), source row-major Bt[N][K]: lane l: n=l&15, kh=l>>4, K=8*kh+e (e<8), 16+8*kh+(e-8) (e>=8)
__device__ __forceinline__ v16h load_fragB(const _Float16* tile, int lane) {
    int n = lane & 15;
    int kh = (lane >> 4) & 1;
    const _Float16* p = tile + n * LDSS + kh * 8;
    v16h r;
    uint4* rp = reinterpret_cast<uint4*>(&r);
    rp[0] = *reinterpret_cast<const uint4*>(p);
    rp[1] = *reinterpret_cast<const uint4*>(p + 16);
    return r;
}
__device__ __forceinline__ v8f wmma_f16(v16h a, v16h b, v8f c) {
    return __builtin_amdgcn_wmma_f32_16x16x32_f16(false, a, false, b, (short)0, c, false, false);
}

// ---------------- generic batched NT GEMM: C[M,N] = A[M,K] * B[N,K]^T (+bias, act) ----------------
// 128 threads = 4 waves (2x2), each wave owns a 32x32 C sub-tile (2x2 WMMA tiles).
// Async double-buffered LDS staging; edge rows are CLAMPED (GEMM outputs only depend on own row/col).
__global__ __launch_bounds__(128)
void gemm_nt_kernel(const _Float16* __restrict__ A, int lda, long long strideA,
                    const _Float16* __restrict__ B, int ldb, long long strideB,
                    float* __restrict__ C, int ldc, long long strideC,
                    int M, int N, int K,
                    const float* __restrict__ bias, int act)
{
    __shared__ _Float16 As[2][64 * LDSS];
    __shared__ _Float16 Bs[2][64 * LDSS];
    const int tid  = threadIdx.x;
    const int lane = tid & 31;
    const int wave = tid >> 5;   // 0..3
    const int wr   = wave >> 1;  // M half
    const int wc   = wave & 1;   // N half
    const int bm = blockIdx.y * 64;
    const int bn = blockIdx.x * 64;
    A += (long long)blockIdx.z * strideA;
    B += (long long)blockIdx.z * strideB;
    C += (long long)blockIdx.z * strideC;

    // per-thread staging chunk coords (2 chunks of 16B per tile per thread)
    int r0 = tid >> 2,            ko0 = (tid & 3) * 8;
    int r1 = (tid + 128) >> 2,    ko1 = ((tid + 128) & 3) * 8;
    int ra0 = bm + r0; if (ra0 >= M) ra0 = M - 1;
    int ra1 = bm + r1; if (ra1 >= M) ra1 = M - 1;
    int rb0 = bn + r0; if (rb0 >= N) rb0 = N - 1;
    int rb1 = bn + r1; if (rb1 >= N) rb1 = N - 1;

    auto stage = [&](int k0, int buf) {
        async_copy_b128(lds_off_of(&As[buf][r0 * LDSS + ko0]), A,
                        (unsigned)(((long long)ra0 * lda + k0 + ko0) * 2));
        async_copy_b128(lds_off_of(&As[buf][r1 * LDSS + ko1]), A,
                        (unsigned)(((long long)ra1 * lda + k0 + ko1) * 2));
        async_copy_b128(lds_off_of(&Bs[buf][r0 * LDSS + ko0]), B,
                        (unsigned)(((long long)rb0 * ldb + k0 + ko0) * 2));
        async_copy_b128(lds_off_of(&Bs[buf][r1 * LDSS + ko1]), B,
                        (unsigned)(((long long)rb1 * ldb + k0 + ko1) * 2));
    };

    v8f acc[2][2] = {};
    const int nk = K >> 5;           // K is always a multiple of 32
    stage(0, 0);
    for (int kt = 0; kt < nk; ++kt) {
        wait_async0();               // own stage-kt copies landed
        __syncthreads();             // all waves landed + done reading other buffer
        if (kt + 1 < nk) stage((kt + 1) * 32, (kt + 1) & 1);
        const _Float16* Ab = As[kt & 1];
        const _Float16* Bb = Bs[kt & 1];
        v16h a0 = load_fragA(Ab + (wr * 32 +  0) * LDSS, lane);
        v16h a1 = load_fragA(Ab + (wr * 32 + 16) * LDSS, lane);
        v16h b0 = load_fragB(Bb + (wc * 32 +  0) * LDSS, lane);
        v16h b1 = load_fragB(Bb + (wc * 32 + 16) * LDSS, lane);
        acc[0][0] = wmma_f16(a0, b0, acc[0][0]);
        acc[0][1] = wmma_f16(a0, b1, acc[0][1]);
        acc[1][0] = wmma_f16(a1, b0, acc[1][0]);
        acc[1][1] = wmma_f16(a1, b1, acc[1][1]);
    }
    // C fragment layout: lane: n = lane&15, mhalf = lane>>4; VGPR r -> row r + 8*mhalf
    const int n_in = lane & 15, mh = lane >> 4;
    #pragma unroll
    for (int mi = 0; mi < 2; ++mi) {
        #pragma unroll
        for (int ni = 0; ni < 2; ++ni) {
            int col = bn + wc * 32 + ni * 16 + n_in;
            if (col >= N) continue;
            float bv = bias ? bias[col] : 0.0f;
            #pragma unroll
            for (int r = 0; r < 8; ++r) {
                int row = bm + wr * 32 + mi * 16 + r + 8 * mh;
                if (row < M) {
                    float v = acc[mi][ni][r] + bv;
                    if (act == 1) v = tanhf(v);
                    C[(long long)row * ldc + col] = v;
                }
            }
        }
    }
}

// ---------------- fused vocab GEMM + per-tile log-sum-exp partials ----------------
// 64x64 logits tile never leaves LDS; emits per-row (max, sumexp) partials + target logit.
__global__ __launch_bounds__(128)
void gemm_lse_kernel(const _Float16* __restrict__ A,   // e_f16 [MROWS, VLAYP]
                     const _Float16* __restrict__ B,   // e2v_f16 [VEN, VLAYP]
                     const float* __restrict__ bias,   // e2v_b [VEN]
                     const int* __restrict__ x_en,     // [BATCH, SEQ]
                     float* __restrict__ pmax, float* __restrict__ psum,
                     float* __restrict__ tlogit)
{
    __shared__ _Float16 As[2][64 * LDSS];
    __shared__ _Float16 Bs[2][64 * LDSS];
    __shared__ float Cs[64][65];
    const int tid  = threadIdx.x;
    const int lane = tid & 31;
    const int wave = tid >> 5;
    const int wr = wave >> 1, wc = wave & 1;
    const int bm = blockIdx.y * 64;   // M tiles exact: 49*64 = 3136
    const int bn = blockIdx.x * 64;   // vocab tiles exact: 500*64 = 32000

    int r0 = tid >> 2,         ko0 = (tid & 3) * 8;
    int r1 = (tid + 128) >> 2, ko1 = ((tid + 128) & 3) * 8;

    auto stage = [&](int k0, int buf) {
        async_copy_b128(lds_off_of(&As[buf][r0 * LDSS + ko0]), A,
                        (unsigned)(((long long)(bm + r0) * VLAYP + k0 + ko0) * 2));
        async_copy_b128(lds_off_of(&As[buf][r1 * LDSS + ko1]), A,
                        (unsigned)(((long long)(bm + r1) * VLAYP + k0 + ko1) * 2));
        async_copy_b128(lds_off_of(&Bs[buf][r0 * LDSS + ko0]), B,
                        (unsigned)(((long long)(bn + r0) * VLAYP + k0 + ko0) * 2));
        async_copy_b128(lds_off_of(&Bs[buf][r1 * LDSS + ko1]), B,
                        (unsigned)(((long long)(bn + r1) * VLAYP + k0 + ko1) * 2));
    };

    v8f acc[2][2] = {};
    const int nk = VLAYP >> 5;
    stage(0, 0);
    for (int kt = 0; kt < nk; ++kt) {
        wait_async0();
        __syncthreads();
        if (kt + 1 < nk) stage((kt + 1) * 32, (kt + 1) & 1);
        const _Float16* Ab = As[kt & 1];
        const _Float16* Bb = Bs[kt & 1];
        v16h a0 = load_fragA(Ab + (wr * 32 +  0) * LDSS, lane);
        v16h a1 = load_fragA(Ab + (wr * 32 + 16) * LDSS, lane);
        v16h b0 = load_fragB(Bb + (wc * 32 +  0) * LDSS, lane);
        v16h b1 = load_fragB(Bb + (wc * 32 + 16) * LDSS, lane);
        acc[0][0] = wmma_f16(a0, b0, acc[0][0]);
        acc[0][1] = wmma_f16(a0, b1, acc[0][1]);
        acc[1][0] = wmma_f16(a1, b0, acc[1][0]);
        acc[1][1] = wmma_f16(a1, b1, acc[1][1]);
    }
    const int n_in = lane & 15, mh = lane >> 4;
    #pragma unroll
    for (int mi = 0; mi < 2; ++mi)
        #pragma unroll
        for (int ni = 0; ni < 2; ++ni) {
            int col = wc * 32 + ni * 16 + n_in;
            float bv = bias[bn + col];
            #pragma unroll
            for (int r = 0; r < 8; ++r)
                Cs[wr * 32 + mi * 16 + r + 8 * mh][col] = acc[mi][ni][r] + bv;
        }
    __syncthreads();
    // per-row reduce over this 64-col tile: 2 threads per row
    int row = tid >> 1, half = tid & 1;
    float mx = -__builtin_inff();
    for (int c2 = half * 32; c2 < half * 32 + 32; ++c2) mx = fmaxf(mx, Cs[row][c2]);
    mx = fmaxf(mx, __shfl_xor(mx, 1));
    float se = 0.0f;
    for (int c2 = half * 32; c2 < half * 32 + 32; ++c2) se += expf(Cs[row][c2] - mx);
    se += __shfl_xor(se, 1);
    if (half == 0) {
        int grow = bm + row;
        pmax[(long long)grow * NTILES_V + blockIdx.x] = mx;
        psum[(long long)grow * NTILES_V + blockIdx.x] = se;
        int b = grow / TM1, t = grow % TM1;
        int y = x_en[b * SEQ + t + 1];
        if (y >= bn && y < bn + 64) tlogit[grow] = Cs[row][y - bn];
    }
}

// combine 500 per-tile partials per row -> reward = target_logit - logsumexp
__global__ void row_lse_kernel(const float* __restrict__ pmax, const float* __restrict__ psum,
                               const float* __restrict__ tlogit, float* __restrict__ reward)
{
    int row = blockIdx.x, lane = threadIdx.x; // 32 threads
    float mx = -__builtin_inff();
    for (int i = lane; i < NTILES_V; i += 32) mx = fmaxf(mx, pmax[(long long)row * NTILES_V + i]);
    for (int m = 16; m; m >>= 1) mx = fmaxf(mx, __shfl_xor(mx, m));
    float s = 0.0f;
    for (int i = lane; i < NTILES_V; i += 32)
        s += psum[(long long)row * NTILES_V + i] * expf(pmax[(long long)row * NTILES_V + i] - mx);
    for (int m = 16; m; m >>= 1) s += __shfl_xor(s, m);
    if (lane == 0) reward[row] = tlogit[row] - (mx + logf(s));
}

// ---------------- elementwise kernels ----------------
__global__ void embed_gather_kernel(const int* __restrict__ tok, const float* __restrict__ emb,
                                    _Float16* __restrict__ out, int n_tok)
{
    long long idx = (long long)blockIdx.x * blockDim.x + threadIdx.x;
    long long total = (long long)n_tok * WDIMP;
    if (idx >= total) return;
    int i = (int)(idx / WDIMP), d = (int)(idx % WDIMP);
    out[idx] = (d < WDIM) ? (_Float16)emb[(long long)tok[i] * WDIM + d] : (_Float16)0.0f;
}

__global__ void f32_to_f16_pad_kernel(const float* __restrict__ src, _Float16* __restrict__ dst,
                                      int M, int K, int Kp)
{
    long long idx = (long long)blockIdx.x * blockDim.x + threadIdx.x;
    long long total = (long long)M * Kp;
    if (idx >= total) return;
    int r = (int)(idx / Kp), k = (int)(idx % Kp);
    dst[idx] = (k < K) ? (_Float16)src[(long long)r * K + k] : (_Float16)0.0f;
}

__global__ void lstm_gate_kernel(const float* __restrict__ xg, int t,
                                 const float* __restrict__ gbuf,
                                 const float* __restrict__ bih, const float* __restrict__ bhh,
                                 float* __restrict__ cbuf,
                                 float* __restrict__ hs32, _Float16* __restrict__ hs16)
{
    int idx = blockIdx.x * blockDim.x + threadIdx.x;
    if (idx >= BATCH * HID) return;
    int b = idx / HID, j = idx % HID;
    const float* xp = xg + ((long long)b * SEQ + t) * GATE;
    const float* gp = gbuf + (long long)b * GATE;
    float gi = xp[j]            + gp[j]            + bih[j]            + bhh[j];
    float gf = xp[j + HID]      + gp[j + HID]      + bih[j + HID]      + bhh[j + HID];
    float gg = xp[j + 2 * HID]  + gp[j + 2 * HID]  + bih[j + 2 * HID]  + bhh[j + 2 * HID];
    float go = xp[j + 3 * HID]  + gp[j + 3 * HID]  + bih[j + 3 * HID]  + bhh[j + 3 * HID];
    float si = 1.0f / (1.0f + expf(-gi));
    float sf = 1.0f / (1.0f + expf(-gf));
    float so = 1.0f / (1.0f + expf(-go));
    float c = sf * cbuf[idx] + si * tanhf(gg);
    float h = so * tanhf(c);
    cbuf[idx] = c;
    hs32[((long long)b * SEQ + t) * HID  + j] = h;
    hs16[((long long)b * SEQ + t) * HIDP + j] = (_Float16)h;
}

__device__ __forceinline__ float u01_hash(unsigned x) {
    x += 0x9e3779b9u; x ^= x >> 16; x *= 0x85ebca6bu; x ^= x >> 13; x *= 0xc2b2ae35u; x ^= x >> 16;
    return ((float)(x >> 8) + 0.5f) * (1.0f / 16777216.0f);
}

// one wave per (b,t): log-softmax over s (S=50) + Gumbel-max sample
__global__ void sample_kernel(const float* __restrict__ scores, int* __restrict__ samples,
                              float* __restrict__ logp)
{
    int bt = blockIdx.x * 8 + (threadIdx.x >> 5);
    if (bt >= MROWS) return;
    int lane = threadIdx.x & 31;
    int b = bt / TM1, t = bt % TM1;
    const float* sc = scores + (long long)b * SEQ * TM1 + t;
    float v0 = (lane      < SEQ) ? sc[(long long)lane * TM1]        : -__builtin_inff();
    float v1 = (lane + 32 < SEQ) ? sc[(long long)(lane + 32) * TM1] : -__builtin_inff();
    float mx = fmaxf(v0, v1);
    for (int m = 16; m; m >>= 1) mx = fmaxf(mx, __shfl_xor(mx, m));
    float se = 0.0f;
    if (lane      < SEQ) se += expf(v0 - mx);
    if (lane + 32 < SEQ) se += expf(v1 - mx);
    for (int m = 16; m; m >>= 1) se += __shfl_xor(se, m);
    float lse = mx + logf(se);
    unsigned base = (unsigned)(bt * 64) ^ 0x2a2a2a2au;
    float z0 = (lane      < SEQ) ? v0 - logf(-logf(u01_hash(base + (unsigned)lane)))        : -__builtin_inff();
    float z1 = (lane + 32 < SEQ) ? v1 - logf(-logf(u01_hash(base + (unsigned)(lane + 32)))) : -__builtin_inff();
    float bz = z0; int bi = lane;
    if (z1 > bz) { bz = z1; bi = lane + 32; }
    for (int m = 16; m; m >>= 1) {
        float oz = __shfl_xor(bz, m);
        int   oi = __shfl_xor(bi, m);
        if (oz > bz || (oz == bz && oi < bi)) { bz = oz; bi = oi; }
    }
    if (lane == 0) {
        samples[bt] = bi;
        logp[bt] = sc[(long long)bi * TM1] - lse;
    }
}

// feat_f16[b*49+t][k] : [dec_h(t) | enc_h(sample) | zero-pad], all f16
__global__ void build_feat_kernel(const float* __restrict__ hs_dec, const float* __restrict__ hs_enc,
                                  const int* __restrict__ samples, _Float16* __restrict__ feat)
{
    long long idx = (long long)blockIdx.x * blockDim.x + threadIdx.x;
    if (idx >= (long long)MROWS * FEATKP) return;
    int row = (int)(idx >> 10), k = (int)(idx & (FEATKP - 1));
    int b = row / TM1, t = row % TM1;
    float v = 0.0f;
    if (k < HID)            v = hs_dec[((long long)b * SEQ + t) * HID + k];
    else if (k < 2 * HID)   v = hs_enc[((long long)b * SEQ + samples[row]) * HID + (k - HID)];
    feat[idx] = (_Float16)v;
}

__global__ void loss_kernel(const int* __restrict__ x_en, const float* __restrict__ reward,
                            const float* __restrict__ logp, float* __restrict__ out)
{
    __shared__ float s_r[TM1], s_rl[TM1];
    int t = threadIdx.x;
    if (t < TM1) {
        const float baseline = logf(1.0f / (float)VEN);
        float sr = 0.0f, srl = 0.0f, cnt = 0.0f;
        for (int b = 0; b < BATCH; ++b) {
            int y = x_en[b * SEQ + t + 1];
            float m = (y != PAD_TOKEN) ? 1.0f : 0.0f;
            int row = b * TM1 + t;
            float r = reward[row];
            sr  += r * m;
            srl += logp[row] * (r - baseline) * m;
            cnt += m;
        }
        cnt = fmaxf(cnt, 1.0f);
        s_r[t] = sr / cnt;
        s_rl[t] = srl / cnt;
    }
    __syncthreads();
    if (t == 0) {
        float a = 0.0f, bsum = 0.0f;
        for (int i = 0; i < TM1; ++i) { a += s_r[i]; bsum += s_rl[i]; }
        out[0] = -a;
        out[1] = -bsum;
    }
}

// ---------------- host ----------------
extern "C" void kernel_launch(void* const* d_in, const int* in_sizes, int n_in,
                              void* d_out, int out_size, void* d_ws, size_t ws_size,
                              hipStream_t stream) {
    (void)in_sizes; (void)n_in; (void)out_size; (void)ws_size;
    const int*   x_de     = (const int*)  d_in[0];
    const int*   x_en     = (const int*)  d_in[1];
    const float* emb_de_w = (const float*)d_in[2];
    const float* emb_en_w = (const float*)d_in[3];
    const float* enc_Wih  = (const float*)d_in[4];
    const float* enc_Whh  = (const float*)d_in[5];
    const float* enc_bih  = (const float*)d_in[6];
    const float* enc_bhh  = (const float*)d_in[7];
    const float* dec_Wih  = (const float*)d_in[8];
    const float* dec_Whh  = (const float*)d_in[9];
    const float* dec_bih  = (const float*)d_in[10];
    const float* dec_bhh  = (const float*)d_in[11];
    const float* h2e_w    = (const float*)d_in[12];
    const float* h2e_b    = (const float*)d_in[13];
    const float* e2v_w    = (const float*)d_in[14];
    const float* e2v_b    = (const float*)d_in[15];
    float* out = (float*)d_out;

    // ---- workspace bump allocator (256B aligned) ----
    char* p = (char*)d_ws;
    auto alloc = [&](size_t n) -> void* { void* r = p; p += (n + 255) & ~(size_t)255; return r; };
    _Float16* wih_e  = (_Float16*)alloc(sizeof(_Float16) * (size_t)GATE * WDIMP);
    _Float16* whh_e  = (_Float16*)alloc(sizeof(_Float16) * (size_t)GATE * HIDP);
    _Float16* wih_d  = (_Float16*)alloc(sizeof(_Float16) * (size_t)GATE * WDIMP);
    _Float16* whh_d  = (_Float16*)alloc(sizeof(_Float16) * (size_t)GATE * HIDP);
    _Float16* h2e_h  = (_Float16*)alloc(sizeof(_Float16) * (size_t)VLAY * FEATKP);
    _Float16* e2v_h  = (_Float16*)alloc(sizeof(_Float16) * (size_t)VEN * VLAYP);
    _Float16* emb_de = (_Float16*)alloc(sizeof(_Float16) * (size_t)BATCH * SEQ * WDIMP);
    _Float16* emb_en = (_Float16*)alloc(sizeof(_Float16) * (size_t)BATCH * SEQ * WDIMP);
    float*    xg_e   = (float*)   alloc(sizeof(float) * (size_t)BATCH * SEQ * GATE);
    float*    xg_d   = (float*)   alloc(sizeof(float) * (size_t)BATCH * SEQ * GATE);
    float*    hs_e32 = (float*)   alloc(sizeof(float) * (size_t)BATCH * SEQ * HID);
    float*    hs_d32 = (float*)   alloc(sizeof(float) * (size_t)BATCH * SEQ * HID);
    _Float16* hs_e16 = (_Float16*)alloc(sizeof(_Float16) * (size_t)BATCH * SEQ * HIDP);
    _Float16* hs_d16 = (_Float16*)alloc(sizeof(_Float16) * (size_t)BATCH * SEQ * HIDP);
    _Float16* h0     = (_Float16*)alloc(sizeof(_Float16) * (size_t)BATCH * HIDP);
    float*    cbuf   = (float*)   alloc(sizeof(float) * (size_t)BATCH * HID);
    float*    gbuf   = (float*)   alloc(sizeof(float) * (size_t)BATCH * GATE);
    float*    scores = (float*)   alloc(sizeof(float) * (size_t)BATCH * SEQ * TM1);
    int*      samples= (int*)     alloc(sizeof(int) * (size_t)MROWS);
    float*    logp   = (float*)   alloc(sizeof(float) * (size_t)MROWS);
    _Float16* feat   = (_Float16*)alloc(sizeof(_Float16) * (size_t)MROWS * FEATKP);
    float*    e32    = (float*)   alloc(sizeof(float) * (size_t)MROWS * VLAY);
    _Float16* e16    = (_Float16*)alloc(sizeof(_Float16) * (size_t)MROWS * VLAYP);
    float*    pmax   = (float*)   alloc(sizeof(float) * (size_t)MROWS * NTILES_V);
    float*    psum   = (float*)   alloc(sizeof(float) * (size_t)MROWS * NTILES_V);
    float*    tlog   = (float*)   alloc(sizeof(float) * (size_t)MROWS);
    float*    reward = (float*)   alloc(sizeof(float) * (size_t)MROWS);

    auto gemm = [&](const _Float16* A, int lda, long long sA,
                    const _Float16* B, int ldb, long long sB,
                    float* C, int ldc, long long sC,
                    int M, int N, int K, const float* bias, int act, int batch) {
        dim3 g((N + 63) / 64, (M + 63) / 64, batch);
        gemm_nt_kernel<<<g, 128, 0, stream>>>(A, lda, sA, B, ldb, sB, C, ldc, sC, M, N, K, bias, act);
    };
    auto cvt = [&](const float* src, _Float16* dst, int M, int K, int Kp) {
        long long total = (long long)M * Kp;
        f32_to_f16_pad_kernel<<<(unsigned)((total + 255) / 256), 256, 0, stream>>>(src, dst, M, K, Kp);
    };

    // ---- convert weights to padded f16 ----
    cvt(enc_Wih, wih_e, GATE, WDIM, WDIMP);
    cvt(enc_Whh, whh_e, GATE, HID, HIDP);
    cvt(dec_Wih, wih_d, GATE, WDIM, WDIMP);
    cvt(dec_Whh, whh_d, GATE, HID, HIDP);
    cvt(h2e_w, h2e_h, VLAY, FEATK, FEATKP);
    cvt(e2v_w, e2v_h, VEN, VLAY, VLAYP);

    // ---- embedding gather (f32 -> padded f16) ----
    {
        long long total = (long long)BATCH * SEQ * WDIMP;
        embed_gather_kernel<<<(unsigned)((total + 255) / 256), 256, 0, stream>>>(x_de, emb_de_w, emb_de, BATCH * SEQ);
        embed_gather_kernel<<<(unsigned)((total + 255) / 256), 256, 0, stream>>>(x_en, emb_en_w, emb_en, BATCH * SEQ);
    }

    // ---- input projections: xg = emb @ Wih^T  (biases added in gate kernel) ----
    gemm(emb_de, WDIMP, 0, wih_e, WDIMP, 0, xg_e, GATE, 0, BATCH * SEQ, GATE, WDIMP, nullptr, 0, 1);
    gemm(emb_en, WDIMP, 0, wih_d, WDIMP, 0, xg_d, GATE, 0, BATCH * SEQ, GATE, WDIMP, nullptr, 0, 1);

    // ---- zero-init padded h buffers ----
    hipMemsetAsync(hs_e16, 0, sizeof(_Float16) * (size_t)BATCH * SEQ * HIDP, stream);
    hipMemsetAsync(hs_d16, 0, sizeof(_Float16) * (size_t)BATCH * SEQ * HIDP, stream);
    hipMemsetAsync(h0, 0, sizeof(_Float16) * (size_t)BATCH * HIDP, stream);

    auto run_lstm = [&](const float* xg, const _Float16* whh, const float* bih, const float* bhh,
                        float* hs32, _Float16* hs16) {
        hipMemsetAsync(cbuf, 0, sizeof(float) * (size_t)BATCH * HID, stream);
        for (int t = 0; t < SEQ; ++t) {
            const _Float16* Aprev = (t == 0) ? h0 : hs16 + (size_t)(t - 1) * HIDP;
            int lda = (t == 0) ? HIDP : SEQ * HIDP;
            gemm(Aprev, lda, 0, whh, HIDP, 0, gbuf, GATE, 0, BATCH, GATE, HIDP, nullptr, 0, 1);
            lstm_gate_kernel<<<(BATCH * HID + 255) / 256, 256, 0, stream>>>(
                xg, t, gbuf, bih, bhh, cbuf, hs32, hs16);
        }
    };
    run_lstm(xg_e, whh_e, enc_bih, enc_bhh, hs_e32, hs_e16);
    run_lstm(xg_d, whh_d, dec_bih, dec_bhh, hs_d32, hs_d16);

    // ---- attention scores: per-batch enc_h[50,H] @ dec_h[:49,H]^T ----
    gemm(hs_e16, HIDP, (long long)SEQ * HIDP,
         hs_d16, HIDP, (long long)SEQ * HIDP,
         scores, TM1, (long long)SEQ * TM1,
         SEQ, TM1, HIDP, nullptr, 0, BATCH);

    // ---- categorical sampling + logp over attention axis ----
    sample_kernel<<<(MROWS + 7) / 8, 256, 0, stream>>>(scores, samples, logp);

    // ---- feat = [dec_h | context] (padded f16) ----
    {
        long long total = (long long)MROWS * FEATKP;
        build_feat_kernel<<<(unsigned)((total + 255) / 256), 256, 0, stream>>>(hs_d32, hs_e32, samples, feat);
    }

    // ---- e = tanh(feat @ h2e^T + b) ----
    gemm(feat, FEATKP, 0, h2e_h, FEATKP, 0, e32, VLAY, 0, MROWS, VLAY, FEATKP, h2e_b, 1, 1);
    cvt(e32, e16, MROWS, VLAY, VLAYP);

    // ---- fused vocab GEMM + LSE partials, then combine ----
    gemm_lse_kernel<<<dim3(NTILES_V, MROWS / 64), 128, 0, stream>>>(e16, e2v_h, e2v_b, x_en, pmax, psum, tlog);
    row_lse_kernel<<<MROWS, 32, 0, stream>>>(pmax, psum, tlog, reward);

    // ---- masked NLL + REINFORCE reduction ----
    loss_kernel<<<1, 64, 0, stream>>>(x_en, reward, logp, out);
}
